// FullAttention_79259326480729
// MI455X (gfx1250) — matrix operand order repaired and checked by reading, not compiled, not run on hardware
//
#include <hip/hip_runtime.h>
#include <hip/hip_bf16.h>

typedef __attribute__((ext_vector_type(16))) _Float16 v16h;
typedef __attribute__((ext_vector_type(8)))  _Float16 v8h;
typedef __attribute__((ext_vector_type(8)))  float    v8f;

// Problem constants (from reference setup_inputs)
#define AN 4
#define AL 2048
#define AS 2048
#define AH 8
#define AD 64
#define BQ 128                  // query rows per workgroup (8 waves x 16)
#define BS 32                   // key/value chunk (WMMA K for f16 is 32)
#define NCH (AS / BS)           // 64 chunks
#define CHUNK_STRIDE (BS * AH * AD)   // floats between consecutive chunks
// fold softmax temperature and log2(e) into Q so v_exp_f32 (native exp2) needs no pre-multiply
#define QSCALE (0.125f * 1.4426950408889634f)

#if __has_builtin(__builtin_amdgcn_sched_barrier)
#define SCHED_FENCE() __builtin_amdgcn_sched_barrier(0)
#else
#define SCHED_FENCE()
#endif

// raw v_exp_f32: base-2, flushes very negative args to 0 (ideal for softmax)
__device__ __forceinline__ float fast_exp2(float x) {
#if __has_builtin(__builtin_amdgcn_exp2f)
    return __builtin_amdgcn_exp2f(x);
#else
    return exp2f(x);
#endif
}

__device__ __forceinline__ v8h shfl_xor16_v8h(v8h x) {
    union { v8h h; int i[4]; } u, r;
    u.h = x;
#pragma unroll
    for (int j = 0; j < 4; ++j) r.i[j] = __shfl_xor(u.i[j], 16, 32);
    return r.h;
}

__device__ __forceinline__ v16h cat8(v8h a, v8h b) {
    return __builtin_shufflevector(a, b, 0,1,2,3,4,5,6,7,8,9,10,11,12,13,14,15);
}

// per-thread register staging for one K/V chunk (double-buffer pipeline)
struct ChunkRegs { float4 k0, k1; float v[8]; };

__device__ __forceinline__ ChunkRegs load_chunk(const float* __restrict__ kptr,
                                                const float* __restrict__ vptr) {
    ChunkRegs r;
    r.k0 = ((const float4*)kptr)[0];
    r.k1 = ((const float4*)kptr)[1];
#pragma unroll
    for (int j = 0; j < 8; ++j) r.v[j] = vptr[(size_t)j * (AH * AD)];
    return r;
}

__device__ __forceinline__ void store_chunk(_Float16* kb, _Float16* vtb,
                                            int tid, const ChunkRegs& r) {
    // K tile row-major [s][d]
    const int s  = tid >> 3;
    const int db = (tid & 7) * 8;
    v8h hk;
    hk[0]=(_Float16)r.k0.x; hk[1]=(_Float16)r.k0.y; hk[2]=(_Float16)r.k0.z; hk[3]=(_Float16)r.k0.w;
    hk[4]=(_Float16)r.k1.x; hk[5]=(_Float16)r.k1.y; hk[6]=(_Float16)r.k1.z; hk[7]=(_Float16)r.k1.w;
    *(v8h*)&kb[s * AD + db] = hk;
    // V tile transposed [d][s]
    const int d  = tid & 63;
    const int sb = (tid >> 6) * 8;
    v8h hv;
#pragma unroll
    for (int j = 0; j < 8; ++j) hv[j] = (_Float16)r.v[j];
    *(v8h*)&vtb[d * BS + sb] = hv;
}

__global__ __launch_bounds__(256)
void flash_attn_f16_wmma(const float* __restrict__ q,
                         const float* __restrict__ k,
                         const float* __restrict__ v,
                         float* __restrict__ out) {
    __shared__ __align__(16) _Float16 kbuf[2][BS * AD];
    __shared__ __align__(16) _Float16 vtbuf[2][AD * BS];

    const int tid  = threadIdx.x;
    const int lane = tid & 31;
    const int wave = tid >> 5;
    const int m    = lane & 15;   // query column index inside 16-tile
    const int hi   = lane >> 4;   // half-wave selector

    const int nQB = AL / BQ;                       // 16
    const int n   = blockIdx.x / (AH * nQB);
    const int h   = (blockIdx.x / nQB) % AH;
    const int qb  = (blockIdx.x % nQB) * BQ + wave * 16;

    const int qrow = qb + m;
    const float* qptr = q + (((size_t)n * AL + qrow) * AH + h) * AD;

    // per-thread global base pointers for the cooperative K/V chunk loads
    const float* kload = k + (((size_t)n * AS + (tid >> 3)) * AH + h) * AD + (tid & 7) * 8;
    const float* vload = v + (((size_t)n * AS + (tid >> 6) * 8) * AH + h) * AD + (tid & 63);

    // ---- Q^T B fragments (32x16 per kstep): lane = column m, halves = d ----
    v16h Bq[2];
#pragma unroll
    for (int ks = 0; ks < 2; ++ks) {
        const int d0 = ks * 32 + hi * 16;          // 16 consecutive d values
        const float4* qv = (const float4*)(qptr + d0);
        float4 f0 = qv[0], f1 = qv[1], f2 = qv[2], f3 = qv[3];
        float tmp[16] = {f0.x,f0.y,f0.z,f0.w, f1.x,f1.y,f1.z,f1.w,
                         f2.x,f2.y,f2.z,f2.w, f3.x,f3.y,f3.z,f3.w};
#pragma unroll
        for (int j = 0; j < 16; ++j) Bq[ks][j] = (_Float16)(tmp[j] * QSCALE);
    }

    // ---- online softmax state (log2 domain) ----
    float m_run = -1e30f;
    float l_run = 0.0f;
    v8f o[4] = {};   // O^T accumulators: 4 d-tiles of 16

    // prologue: stage + commit chunk 0 into buffer 0
    {
        ChunkRegs c0r = load_chunk(kload, vload);
        store_chunk(kbuf[0], vtbuf[0], tid, c0r);
    }
    __syncthreads();

    for (int i = 0; i < NCH; ++i) {
        const _Float16* kb  = kbuf[i & 1];
        const _Float16* vtb = vtbuf[i & 1];
        _Float16* kbn  = kbuf[(i + 1) & 1];
        _Float16* vtbn = vtbuf[(i + 1) & 1];

        // 1) issue global loads for chunk i+1 (latency hidden behind compute)
        ChunkRegs nxt;
        const bool have_next = (i + 1) < NCH;
        if (have_next)
            nxt = load_chunk(kload + (size_t)(i + 1) * CHUNK_STRIDE,
                             vload + (size_t)(i + 1) * CHUNK_STRIDE);

        // 2) all four K A-fragments issued as one DS clause, THEN the WMMAs.
        //    sched_barrier stops the scheduler from re-sinking each load pair
        //    to just before its consumer (round-2 asm showed s_wait_dscnt 0
        //    in front of every WMMA).
        v16h aK[4];
#pragma unroll
        for (int ks = 0; ks < 2; ++ks)
#pragma unroll
            for (int st = 0; st < 2; ++st) {
                const int base = (st * 16 + m) * AD + ks * 32 + hi * 16;
                aK[ks * 2 + st] = cat8(*(const v8h*)&kb[base],
                                       *(const v8h*)&kb[base + 8]);
            }
        SCHED_FENCE();

        v8f c0 = {}, c1 = {};
        c0 = __builtin_amdgcn_wmma_f32_16x16x32_f16(false, aK[0], false, Bq[0], (short)0, c0, false, false);
        c1 = __builtin_amdgcn_wmma_f32_16x16x32_f16(false, aK[1], false, Bq[0], (short)0, c1, false, false);
        c0 = __builtin_amdgcn_wmma_f32_16x16x32_f16(false, aK[2], false, Bq[1], (short)0, c0, false, false);
        c1 = __builtin_amdgcn_wmma_f32_16x16x32_f16(false, aK[3], false, Bq[1], (short)0, c1, false, false);
        SCHED_FENCE();

        // 3) V^T A-fragment loads here: independent DS ops fill the
        //    WMMA->VALU hazard slots and their latency hides under softmax
        v16h aV[4];
#pragma unroll
        for (int dt = 0; dt < 4; ++dt) {
            const int vb = (dt * 16 + m) * BS + hi * 16;
            aV[dt] = cat8(*(const v8h*)&vtb[vb],
                          *(const v8h*)&vtb[vb + 8]);
        }
        SCHED_FENCE();

        // 4) online softmax over this chunk's 32 keys (log2 domain, raw v_exp_f32)
        float mx = -1e30f;
#pragma unroll
        for (int r = 0; r < 8; ++r) mx = fmaxf(mx, fmaxf(c0[r], c1[r]));
        mx = fmaxf(mx, __shfl_xor(mx, 16, 32));
        const float mnew  = fmaxf(m_run, mx);
        const float scale = fast_exp2(m_run - mnew);
        m_run = mnew;

        float p0[8], p1[8], ls = 0.0f;
#pragma unroll
        for (int r = 0; r < 8; ++r) {
            p0[r] = fast_exp2(c0[r] - mnew);
            p1[r] = fast_exp2(c1[r] - mnew);
            ls += p0[r] + p1[r];
        }
        ls += __shfl_xor(ls, 16, 32);
        l_run = l_run * scale + ls;
#pragma unroll
        for (int dt = 0; dt < 4; ++dt)
#pragma unroll
            for (int r = 0; r < 8; ++r) o[dt][r] *= scale;

        // 5) build P^T B fragment (32x16) from C-layout scores via xor16
        v8h ph0, ph1;
#pragma unroll
        for (int r = 0; r < 8; ++r) { ph0[r] = (_Float16)p0[r]; ph1[r] = (_Float16)p1[r]; }
        v8h ot0 = shfl_xor16_v8h(ph0);   // partner rows s=8..15
        v8h ot1 = shfl_xor16_v8h(ph1);   // partner rows s=16..23
        v16h Bp = hi ? cat8(ot1, ph1)    // k = s 16..31
                     : cat8(ph0, ot0);   // k = s  0..15

        // 6) O^T += V^T . P^T : 4 d-tiles, K-dim = 32 keys
        o[0] = __builtin_amdgcn_wmma_f32_16x16x32_f16(false, aV[0], false, Bp, (short)0, o[0], false, false);
        o[1] = __builtin_amdgcn_wmma_f32_16x16x32_f16(false, aV[1], false, Bp, (short)0, o[1], false, false);
        o[2] = __builtin_amdgcn_wmma_f32_16x16x32_f16(false, aV[2], false, Bp, (short)0, o[2], false, false);
        o[3] = __builtin_amdgcn_wmma_f32_16x16x32_f16(false, aV[3], false, Bp, (short)0, o[3], false, false);

        // 7) commit staged chunk i+1 into the other buffer; single barrier per iter
        if (have_next)
            store_chunk(kbn, vtbn, tid, nxt);
        __syncthreads();
    }

    // ---- epilogue: normalize and store (8 contiguous floats per fragment) ----
    const float rl = 1.0f / l_run;
    float* op = out + (((size_t)n * AL + qrow) * AH + h) * AD;
#pragma unroll
    for (int dt = 0; dt < 4; ++dt) {
        float t[8];
#pragma unroll
        for (int r = 0; r < 8; ++r) t[r] = o[dt][r] * rl;
        float4* dst = (float4*)&op[dt * 16 + hi * 8];
        dst[0] = make_float4(t[0], t[1], t[2], t[3]);
        dst[1] = make_float4(t[4], t[5], t[6], t[7]);
    }
}

extern "C" void kernel_launch(void* const* d_in, const int* in_sizes, int n_in,
                              void* d_out, int out_size, void* d_ws, size_t ws_size,
                              hipStream_t stream) {
    const float* q = (const float*)d_in[0];
    const float* k = (const float*)d_in[1];
    const float* v = (const float*)d_in[2];
    float* out = (float*)d_out;
    (void)in_sizes; (void)n_in; (void)out_size; (void)d_ws; (void)ws_size;

    dim3 grid(AN * AH * (AL / BQ));   // 4*8*16 = 512 workgroups
    dim3 block(256);                   // 8 wave32
    flash_attn_f16_wmma<<<grid, block, 0, stream>>>(q, k, v, out);
}
